// Attention2_40261023433212
// MI455X (gfx1250) — matrix operand order repaired and checked by reading, not compiled, not run on hardware
//
#include <hip/hip_runtime.h>

typedef __attribute__((ext_vector_type(16))) _Float16 v16h;
typedef __attribute__((ext_vector_type(8)))  float    v8f;

#define Hs    160
#define Ws    160
#define Cs    64
#define Lg    79             // 79x79 patch grid (stride-2, k=5, pad=1)
#define L     (Lg*Lg)        // 6241 patches
#define Pdim  1600           // 64*5*5 patch feature length
#define KCH   50             // Pdim / 32 k-chunks
#define NT2   392            // tile count padded to even (ceil(6241/16)=391 -> 392)
#define LPAD2 (NT2*16)       // 6272
#define MT2   (NT2/2)        // 196 m-tile pairs
#define GG    14             // l-chunk groups
#define TPG2  28             // tiles per group (14*28 = 392), even

// ---------- 1) per-patch 1/norm (one wave32 per patch) ----------
__global__ void norms_kernel(const float* __restrict__ fk, const float* __restrict__ fq,
                             float* __restrict__ rnK, float* __restrict__ rnQ) {
    int bid  = blockIdx.x;
    int lane = threadIdx.x;
    bool isQ = bid >= L;
    int l    = isQ ? bid - L : bid;
    const float* src = isQ ? fq : fk;
    int ly = l / Lg, lx = l % Lg;
    float acc = 0.f;
    for (int p = lane; p < Pdim; p += 32) {
        int c = p / 25, rem = p % 25;
        int kh = rem / 5, kw = rem % 5;
        int h = ly * 2 - 1 + kh, w = lx * 2 - 1 + kw;
        float v = 0.f;
        if ((unsigned)h < (unsigned)Hs && (unsigned)w < (unsigned)Ws)
            v = src[(c * Hs + h) * Ws + w];
        acc += v * v;
    }
    for (int off = 16; off; off >>= 1) acc += __shfl_xor(acc, off, 32);
    if (lane == 0) {
        float r = 1.f / fmaxf(sqrtf(acc), 1e-12f);
        (isQ ? rnQ : rnK)[l] = r;
    }
}

// ---------- 2) unfold + normalize + f16 + swizzle into WMMA register image ----------
// Apack/Bpack element [ (tile*KCH + kc)*32 + lane ] is the v16h this lane feeds to wmma.
__global__ void pack_kernel(const float* __restrict__ fk, const float* __restrict__ fq,
                            const float* __restrict__ rnK, const float* __restrict__ rnQ,
                            _Float16* __restrict__ Apack, _Float16* __restrict__ Bpack) {
    int tid = blockIdx.x * blockDim.x + threadIdx.x;
    const int per = NT2 * KCH * 32;
    if (tid >= 2 * per) return;
    bool isB = tid >= per;
    int t    = isB ? tid - per : tid;
    int tile = t / (KCH * 32);
    int kc   = (t / 32) % KCH;
    int lane = t % 32;

    v16h out;
    if (!isB) {
        // A (keys, feat_edit): 16x32 f16 operand layout (ISA 7.12.2)
        int l = tile * 16 + (lane & 15);
        float rn = (l < L) ? rnK[l] : 0.f;
        int ly = (l < L) ? (l / Lg) : 0;
        int lx = (l < L) ? (l % Lg) : 0;
        #pragma unroll
        for (int e = 0; e < 16; ++e) {
            int K = (lane < 16) ? ((e < 8) ? e : e + 8)
                                : ((e < 8) ? e + 8 : e + 16);
            int p = kc * 32 + K;
            int c = p / 25, rem = p % 25, kh = rem / 5, kw = rem % 5;
            int h = ly * 2 - 1 + kh, w = lx * 2 - 1 + kw;
            float v = 0.f;
            if (l < L && (unsigned)h < (unsigned)Hs && (unsigned)w < (unsigned)Ws)
                v = fk[(c * Hs + h) * Ws + w];
            out[e] = (_Float16)(v * rn);
        }
        ((v16h*)Apack)[(size_t)(tile * KCH + kc) * 32 + lane] = out;
    } else {
        // B (queries, feat_ori): 32x16 operand layout, lane = column, halves = K
        int m = tile * 16 + (lane & 15);
        float rn = (m < L) ? rnQ[m] : 0.f;
        int my = (m < L) ? (m / Lg) : 0;
        int mx = (m < L) ? (m % Lg) : 0;
        #pragma unroll
        for (int e = 0; e < 16; ++e) {
            int K = (lane < 16) ? ((e < 8) ? e : e + 8)
                                : ((e < 8) ? e + 8 : e + 16);
            int p = kc * 32 + K;
            int c = p / 25, rem = p % 25, kh = rem / 5, kw = rem % 5;
            int h = my * 2 - 1 + kh, w = mx * 2 - 1 + kw;
            float v = 0.f;
            if (m < L && (unsigned)h < (unsigned)Hs && (unsigned)w < (unsigned)Ws)
                v = fq[(c * Hs + h) * Ws + w];
            out[e] = (_Float16)(v * rn);
        }
        ((v16h*)Bpack)[(size_t)(tile * KCH + kc) * 32 + lane] = out;
    }
}

// ---------- 3) fused GEMM + running column-max, 2x2 tile blocking per wave ----------
__global__ void gemm_max_kernel(const _Float16* __restrict__ Apack,
                                const _Float16* __restrict__ Bpack,
                                float* __restrict__ Rpart) {
    int mp   = blockIdx.x;           // m-tile pair: columns of 2 query tiles
    int g    = blockIdx.y;           // l-chunk group (key tiles)
    int lane = threadIdx.x;
    const v16h* Ap = (const v16h*)Apack;
    const v16h* Bp = (const v16h*)Bpack;

    int mt0 = mp * 2, mt1 = mt0 + 1;
    const v16h* b0row = Bp + (size_t)mt0 * KCH * 32 + lane;
    const v16h* b1row = Bp + (size_t)mt1 * KCH * 32 + lane;

    float max0[8], max1[8];
    #pragma unroll
    for (int r = 0; r < 8; ++r) { max0[r] = -INFINITY; max1[r] = -INFINITY; }
    int rowOff = (lane < 16) ? 0 : 8;

    int lt0 = g * TPG2;
    for (int lt = lt0; lt < lt0 + TPG2; lt += 2) {
        const v16h* a0row = Ap + (size_t)lt * KCH * 32 + lane;
        const v16h* a1row = a0row + KCH * 32;
        __builtin_prefetch(a1row + KCH * 32, 0, 0);   // next l-tile pair (global_prefetch_b8)
        v8f acc00 = {}, acc01 = {}, acc10 = {}, acc11 = {};
        for (int kc = 0; kc < KCH; ++kc) {
            v16h a0 = a0row[kc * 32];
            v16h a1 = a1row[kc * 32];
            v16h b0 = b0row[kc * 32];
            v16h b1 = b1row[kc * 32];
            acc00 = __builtin_amdgcn_wmma_f32_16x16x32_f16(false, a0, false, b0, (short)0, acc00, false, false);
            acc01 = __builtin_amdgcn_wmma_f32_16x16x32_f16(false, a0, false, b1, (short)0, acc01, false, false);
            acc10 = __builtin_amdgcn_wmma_f32_16x16x32_f16(false, a1, false, b0, (short)0, acc10, false, false);
            acc11 = __builtin_amdgcn_wmma_f32_16x16x32_f16(false, a1, false, b1, (short)0, acc11, false, false);
        }
        int rb0 = lt * 16 + rowOff;                   // C layout: VGPR r -> M = r (+8 hi lanes)
        int rb1 = rb0 + 16;
        #pragma unroll
        for (int r = 0; r < 8; ++r) {
            if (rb0 + r < L) {
                max0[r] = fmaxf(max0[r], acc00[r]);
                max1[r] = fmaxf(max1[r], acc01[r]);
            }
            if (rb1 + r < L) {
                max0[r] = fmaxf(max0[r], acc10[r]);
                max1[r] = fmaxf(max1[r], acc11[r]);
            }
        }
    }
    float mx0 = max0[0], mx1 = max1[0];
    #pragma unroll
    for (int r = 1; r < 8; ++r) { mx0 = fmaxf(mx0, max0[r]); mx1 = fmaxf(mx1, max1[r]); }
    mx0 = fmaxf(mx0, __shfl_xor(mx0, 16, 32));        // combine M=0..7 with M=8..15 lanes
    mx1 = fmaxf(mx1, __shfl_xor(mx1, 16, 32));
    if (lane < 16) {
        Rpart[(size_t)g * LPAD2 + mt0 * 16 + lane] = mx0;
        Rpart[(size_t)g * LPAD2 + mt1 * 16 + lane] = mx1;
    }
}

// ---------- 4) combine chunk partials ----------
__global__ void reduce_kernel(const float* __restrict__ Rpart, float* __restrict__ outB) {
    int m = blockIdx.x * blockDim.x + threadIdx.x;
    if (m >= L) return;
    float mx = -INFINITY;
    #pragma unroll
    for (int g = 0; g < GG; ++g) mx = fmaxf(mx, Rpart[(size_t)g * LPAD2 + m]);
    outB[m] = mx;
}

extern "C" void kernel_launch(void* const* d_in, const int* in_sizes, int n_in,
                              void* d_out, int out_size, void* d_ws, size_t ws_size,
                              hipStream_t stream) {
    const float* feat_edit = (const float*)d_in[0];   // keys
    const float* feat_ori  = (const float*)d_in[1];   // queries
    float* out = (float*)d_out;                       // (4,1,79,79)

    char* ws = (char*)d_ws;
    size_t off = 0;
    auto alloc = [&](size_t bytes) {
        void* p = ws + off;
        off += (bytes + 255) & ~(size_t)255;
        return p;
    };
    float*    rnK   = (float*)alloc((size_t)LPAD2 * 4);
    float*    rnQ   = (float*)alloc((size_t)LPAD2 * 4);
    _Float16* Apack = (_Float16*)alloc((size_t)NT2 * KCH * 32 * 16 * 2);
    _Float16* Bpack = (_Float16*)alloc((size_t)NT2 * KCH * 32 * 16 * 2);
    float*    Rpart = (float*)alloc((size_t)GG * LPAD2 * 4);
    (void)ws_size; (void)in_sizes; (void)n_in; (void)out_size;

    const size_t featsz = (size_t)Cs * Hs * Ws;
    for (int b = 0; b < 4; ++b) {                     // serialize batches: ws stays ~40.6 MB
        const float* fk = feat_edit + b * featsz;
        const float* fq = feat_ori  + b * featsz;
        norms_kernel<<<2 * L, 32, 0, stream>>>(fk, fq, rnK, rnQ);
        int items = 2 * NT2 * KCH * 32;
        pack_kernel<<<(items + 255) / 256, 256, 0, stream>>>(fk, fq, rnK, rnQ, Apack, Bpack);
        dim3 grid(MT2, GG);
        gemm_max_kernel<<<grid, 32, 0, stream>>>(Apack, Bpack, Rpart);
        reduce_kernel<<<(L + 255) / 256, 256, 0, stream>>>(Rpart, out + (size_t)b * L);
    }
}
